// RCNN_31585189495231
// MI455X (gfx1250) — compile-verified, matrix operand order checked by reading, not run on hardware
//
#include <hip/hip_runtime.h>
#include <hip/hip_bf16.h>
#include <stdint.h>
#include <stddef.h>

typedef _Float16 f16;
typedef _Float16 v8h  __attribute__((ext_vector_type(8)));
typedef _Float16 v16h __attribute__((ext_vector_type(16)));
typedef float    v8f  __attribute__((ext_vector_type(8)));

static __device__ __forceinline__ v16h cat16(v8h lo, v8h hi) {
  return __builtin_shufflevector(lo, hi, 0,1,2,3,4,5,6,7,8,9,10,11,12,13,14,15);
}

// ---------------------------------------------------------------- utilities
__global__ void k_zero(uint32_t* __restrict__ p, size_t n) {
  size_t i = (size_t)blockIdx.x * blockDim.x + threadIdx.x;
  size_t s = (size_t)gridDim.x * blockDim.x;
  for (; i < n; i += s) p[i] = 0u;
}

// w f32 [oc][ic][3][3]  ->  o f16 [tap][oc][ic]
__global__ void k_wtrans(const float* __restrict__ w, f16* __restrict__ o) {
  int e = blockIdx.x * blockDim.x + threadIdx.x;
  if (e >= 9 * 128 * 128) return;
  int ic = e & 127, oc = (e >> 7) & 127, tap = e >> 14;
  o[e] = (f16)w[(oc * 128 + ic) * 9 + tap];
}

// ---------------------------------------------------------------- conv1 path
__device__ __forceinline__ float conv1_val(const float* __restrict__ x,
                                           const float* __restrict__ w,
                                           const float* __restrict__ bias,
                                           int b, int h, int ww, int oc) {
  float a = bias[oc];
  for (int ic = 0; ic < 3; ++ic) {
    const float* xp = x + ((size_t)(b * 3 + ic) << 14);  // 128*128
    const float* wp = w + (oc * 3 + ic) * 25;
#pragma unroll
    for (int dy = 0; dy < 5; ++dy) {
      int iy = h + dy - 2;
      if ((unsigned)iy >= 128u) continue;
#pragma unroll
      for (int dx = 0; dx < 5; ++dx) {
        int ix = ww + dx - 2;
        if ((unsigned)ix >= 128u) continue;
        a += xp[iy * 128 + ix] * wp[dy * 5 + dx];
      }
    }
  }
  return a;
}

__global__ void k_conv1_stats(const float* __restrict__ x, const float* __restrict__ w,
                              const float* __restrict__ bias,
                              float* __restrict__ sum, float* __restrict__ sumsq) {
  __shared__ float ls[128], lq[128];
  int tid = threadIdx.x;
  if (tid < 128) { ls[tid] = 0.f; lq[tid] = 0.f; }
  __syncthreads();
  const size_t total = (size_t)32 * 128 * 128 * 128;
  size_t stride = (size_t)gridDim.x * blockDim.x;   // multiple of 128
  float s = 0.f, q = 0.f;
  for (size_t e = (size_t)blockIdx.x * blockDim.x + tid; e < total; e += stride) {
    int oc = (int)(e & 127);
    int ox = (int)((e >> 7) & 127);
    int oy = (int)((e >> 14) & 127);
    int b  = (int)(e >> 21);
    float a = conv1_val(x, w, bias, b, oy, ox, oc);
    s += a; q += a * a;
  }
  atomicAdd(&ls[tid & 127], s);
  atomicAdd(&lq[tid & 127], q);
  __syncthreads();
  if (tid < 128) { atomicAdd(&sum[tid], ls[tid]); atomicAdd(&sumsq[tid], lq[tid]); }
}

// conv1 + BN + ReLU + 3x3 s2 maxpool, write into padded 65x65 interior
__global__ void k_conv1_bn_pool(const float* __restrict__ x, const float* __restrict__ w,
                                const float* __restrict__ bias,
                                const float* __restrict__ scale, const float* __restrict__ shift,
                                f16* __restrict__ out) {
  size_t e = (size_t)blockIdx.x * blockDim.x + threadIdx.x;
  const size_t total = (size_t)32 * 63 * 63 * 128;
  if (e >= total) return;
  int oc = (int)(e & 127);
  int px = (int)((e >> 7) % 63);
  int py = (int)((e / (128 * 63)) % 63);
  int b  = (int)(e / ((size_t)128 * 63 * 63));
  float sc = scale[oc], sh = shift[oc];
  float m = 0.f;  // relu outputs are >= 0
  for (int dy = 0; dy < 3; ++dy)
    for (int dx = 0; dx < 3; ++dx) {
      float v = conv1_val(x, w, bias, b, 2 * py + dy, 2 * px + dx, oc);
      v = sc * v + sh;
      v = v > 0.f ? v : 0.f;
      m = v > m ? v : m;
    }
  out[(((size_t)b * 65 + py + 1) * 65 + px + 1) * 128 + oc] = (f16)m;
}

// ---------------------------------------------------------------- BN helpers
__global__ void k_chan_stats(const f16* __restrict__ t, float* __restrict__ sum,
                             float* __restrict__ sumsq, size_t total) {
  __shared__ float ls[128], lq[128];
  int tid = threadIdx.x;
  if (tid < 128) { ls[tid] = 0.f; lq[tid] = 0.f; }
  __syncthreads();
  size_t stride = (size_t)gridDim.x * blockDim.x;   // multiple of 128
  float s = 0.f, q = 0.f;
  for (size_t e = (size_t)blockIdx.x * blockDim.x + tid; e < total; e += stride) {
    float v = (float)t[e];
    s += v; q += v * v;
  }
  atomicAdd(&ls[tid & 127], s);
  atomicAdd(&lq[tid & 127], q);
  __syncthreads();
  if (tid < 128) { atomicAdd(&sum[tid], ls[tid]); atomicAdd(&sumsq[tid], lq[tid]); }
}

__global__ void k_finalize_stats(float* __restrict__ sum, float* __restrict__ sumsq,
                                 const float* __restrict__ g, const float* __restrict__ bta,
                                 float* __restrict__ scale, float* __restrict__ shift,
                                 float inv_cnt) {
  int c = threadIdx.x;
  float m = sum[c] * inv_cnt;
  float v = sumsq[c] * inv_cnt - m * m;   // biased variance
  float sc = g[c] * rsqrtf(v + 1e-5f);
  scale[c] = sc;
  shift[c] = bta[c] - m * sc;
  sum[c] = 0.f; sumsq[c] = 0.f;           // ready for next BN
}

// affine + relu, write into padded buffer interior
__global__ void k_bn_relu_pad(const f16* __restrict__ t, const float* __restrict__ scale,
                              const float* __restrict__ shift, f16* __restrict__ out,
                              int H, size_t total) {
  size_t e = (size_t)blockIdx.x * blockDim.x + threadIdx.x;
  if (e >= total) return;
  int ch = (int)(e & 127);
  size_t r = e >> 7;
  int x = (int)(r % H); r /= H;
  int y = (int)(r % H);
  int b = (int)(r / H);
  float v = scale[ch] * (float)t[e] + shift[ch];
  v = v > 0.f ? v : 0.f;
  int Wp = H + 2;
  out[(((size_t)b * Wp + y + 1) * Wp + x + 1) * 128 + ch] = (f16)v;
}

// ---------------------------------------------------------------- WMMA conv3x3
// in:  padded NHWC f16 [B, H+2, H+2, 128] with zero halo
// wgt: f16 [tap(9)][oc(128)][ic(128)]
// out: unpadded NHWC f16 [B, H, H, 128]; out = conv + bias (+ res)
// Per wave: 32 output pixels (two 16-px M tiles) x 64 output channels.
__global__ __launch_bounds__(128) void k_conv3x3_wmma(
    const f16* __restrict__ in, const f16* __restrict__ wgt,
    const float* __restrict__ bias, const f16* __restrict__ res,
    f16* __restrict__ out, int H) {
  const int W = H, Wp = H + 2;
  const int lane = threadIdx.x & 31;
  const int wave = threadIdx.x >> 5;
  const int xsegs = (W + 31) >> 5;             // 32-pixel segments
  const int tile = blockIdx.x * 4 + wave;
  if (tile >= 32 * H * xsegs) return;          // wave-uniform exit; EXEC stays full
  const int xseg = tile % xsegs;
  const int t2 = tile / xsegs;
  const int y = t2 % H;
  const int b = t2 / H;
  const int n0 = blockIdx.y * 64;
  const int col = lane & 15;                   // M (A) / N (B,D)
  const int g = lane >> 4;                     // K-half selector
  const int px0 = xseg * 32 + col;             // first M-tile pixel

  v8f acc[2][4] = {};                          // [m-half][oc subtile]

  const size_t obase = ((size_t)b * H + y) * (size_t)W * 128;
  if (res) {
    // residual row is consumed only after the K loop; warm it into cache
    __builtin_prefetch((const void*)(res + obase + (size_t)(xseg * 32) * 128), 0, 3);
  }

  const f16* inb = in + (size_t)b * (H + 2) * Wp * 128;
  for (int ky = 0; ky < 3; ++ky) {
    const f16* rowp = inb + ((size_t)(y + ky) * Wp + px0) * 128;
    for (int kx = 0; kx < 3; ++kx) {
      const f16* p = rowp + kx * 128;
      const f16* wt = wgt + (size_t)(ky * 3 + kx) * 16384 + (size_t)(n0 + col) * 128;
#pragma unroll
      for (int kc = 0; kc < 4; ++kc) {
        // A fragments: lane (M=col, half g) -> two contiguous 16B runs of K
        const f16* ap = p + kc * 32 + g * 8;
        v16h A0 = cat16(*(const v8h*)ap, *(const v8h*)(ap + 16));
        const f16* ap1 = ap + 16 * 128;        // +16 pixels
        v16h A1 = cat16(*(const v8h*)ap1, *(const v8h*)(ap1 + 16));
        // B fragments: lane (N=oc, half g) -> same pattern in [tap][oc][ic]
        const f16* bp = wt + kc * 32 + g * 8;
#pragma unroll
        for (int j = 0; j < 4; ++j) {
          v16h Bj = cat16(*(const v8h*)bp, *(const v8h*)(bp + 16));
          acc[0][j] = __builtin_amdgcn_wmma_f32_16x16x32_f16(
              false, A0, false, Bj, (short)0, acc[0][j], false, false);
          acc[1][j] = __builtin_amdgcn_wmma_f32_16x16x32_f16(
              false, A1, false, Bj, (short)0, acc[1][j], false, false);
          bp += 16 * 128;
        }
      }
    }
  }

  // D layout: lane holds channel n0+j*16+col, pixels v + g*8 (v = VGPR index)
#pragma unroll
  for (int m = 0; m < 2; ++m) {
#pragma unroll
    for (int j = 0; j < 4; ++j) {
      const int ch = n0 + j * 16 + col;
      const float bb = bias[ch];
#pragma unroll
      for (int v = 0; v < 8; ++v) {
        const int x = xseg * 32 + m * 16 + g * 8 + v;
        if (x < W) {
          const size_t idx = obase + (size_t)x * 128 + ch;
          float val = acc[m][j][v] + bb;
          if (res) val += (float)res[idx];
          out[idx] = (f16)val;
        }
      }
    }
  }
}

// ---------------------------------------------------------------- pool / head
__global__ void k_maxpool(const f16* __restrict__ in, f16* __restrict__ out,
                          int Hin, int Hout, int inPad, int outPad, size_t total) {
  size_t e = (size_t)blockIdx.x * blockDim.x + threadIdx.x;
  if (e >= total) return;
  int c = (int)(e & 127);
  size_t r = e >> 7;
  int ox = (int)(r % Hout); r /= Hout;
  int oy = (int)(r % Hout);
  int b = (int)(r / Hout);
  int Wi = Hin + 2 * inPad, Wo = Hout + 2 * outPad;
  float m = -1e30f;
  for (int dy = 0; dy < 3; ++dy)
    for (int dx = 0; dx < 3; ++dx) {
      int iy = 2 * oy + dy + inPad, ix = 2 * ox + dx + inPad;
      float v = (float)in[(((size_t)b * Wi + iy) * Wi + ix) * 128 + c];
      m = v > m ? v : m;
    }
  out[(((size_t)b * Wo + oy + outPad) * Wo + ox + outPad) * 128 + c] = (f16)m;
}

__global__ void k_gmax(const f16* __restrict__ p, float* __restrict__ gm) {
  int tid = blockIdx.x * blockDim.x + threadIdx.x;
  if (tid >= 4096) return;
  int b = tid >> 7, c = tid & 127;
  const f16* q = p + (size_t)b * 225 * 128 + c;
  float m = -1e30f;
  for (int i = 0; i < 225; ++i) {
    float v = (float)q[(size_t)i * 128];
    m = v > m ? v : m;
  }
  gm[tid] = m;
}

__global__ void k_fc(const float* __restrict__ gm, const float* __restrict__ w,
                     const float* __restrict__ bb, float* __restrict__ out) {
  int tid = blockIdx.x * blockDim.x + threadIdx.x;
  if (tid >= 320) return;
  int b = tid / 10, j = tid % 10;
  float a = bb[j];
  const float* gv = gm + b * 128;
  const float* wv = w + j * 128;
  for (int c = 0; c < 128; ++c) a += gv[c] * wv[c];
  out[tid] = a;
}

// ---------------------------------------------------------------- launcher
extern "C" void kernel_launch(void* const* d_in, const int* in_sizes, int n_in,
                              void* d_out, int out_size, void* d_ws, size_t ws_size,
                              hipStream_t stream) {
  (void)in_sizes; (void)n_in; (void)out_size; (void)ws_size;

  const float* X   = (const float*)d_in[0];
  const float* W1  = (const float*)d_in[1];
  const float* B1  = (const float*)d_in[2];
  const float* G0  = (const float*)d_in[3];
  const float* Bt0 = (const float*)d_in[4];
  const float* cfw[4] = {(const float*)d_in[5],  (const float*)d_in[11],
                         (const float*)d_in[17], (const float*)d_in[23]};
  const float* cfb[4] = {(const float*)d_in[6],  (const float*)d_in[12],
                         (const float*)d_in[18], (const float*)d_in[24]};
  const float* crw[4] = {(const float*)d_in[7],  (const float*)d_in[13],
                         (const float*)d_in[19], (const float*)d_in[25]};
  const float* crb[4] = {(const float*)d_in[8],  (const float*)d_in[14],
                         (const float*)d_in[20], (const float*)d_in[26]};
  const float* gg[4]  = {(const float*)d_in[9],  (const float*)d_in[15],
                         (const float*)d_in[21], (const float*)d_in[27]};
  const float* bb[4]  = {(const float*)d_in[10], (const float*)d_in[16],
                         (const float*)d_in[22], (const float*)d_in[28]};
  const float* FCW = (const float*)d_in[29];
  const float* FCB = (const float*)d_in[30];
  float* OUT = (float*)d_out;

  // workspace layout
  size_t off = 0;
  auto alloc = [&](size_t bytes) -> char* {
    char* p = (char*)d_ws + off;
    off = (off + bytes + 255) & ~(size_t)255;
    return p;
  };
  f16* wgt[8];
  for (int i = 0; i < 8; ++i) wgt[i] = (f16*)alloc(9 * 128 * 128 * 2);  // 294912 B each
  float* sum   = (float*)alloc(512);
  float* sumsq = (float*)alloc(512);
  float* scale = (float*)alloc(512);
  float* shift = (float*)alloc(512);
  const size_t PAD63 = (size_t)32 * 65 * 65 * 128;       // elems
  const size_t PAD31 = (size_t)32 * 33 * 33 * 128;
  f16* padA = (f16*)alloc(PAD63 * 2 + 8192);
  f16* padB = (f16*)alloc(PAD63 * 2 + 8192);
  f16* padC = (f16*)alloc(PAD31 * 2 + 8192);
  f16* padD = (f16*)alloc(PAD31 * 2 + 8192);
  f16* f63  = (f16*)alloc((size_t)32 * 63 * 63 * 128 * 2);
  f16* t63  = (f16*)alloc((size_t)32 * 63 * 63 * 128 * 2);
  f16* f31  = (f16*)alloc((size_t)32 * 31 * 31 * 128 * 2);
  f16* t31  = (f16*)alloc((size_t)32 * 31 * 31 * 128 * 2);
  f16* p15  = (f16*)alloc((size_t)32 * 15 * 15 * 128 * 2);
  float* gm = (float*)alloc(32 * 128 * 4);

  // zero halo buffers + stats (every call: harness poisons ws)
  k_zero<<<4096, 256, 0, stream>>>((uint32_t*)padA, (PAD63 * 2 + 8192) / 4);
  k_zero<<<4096, 256, 0, stream>>>((uint32_t*)padB, (PAD63 * 2 + 8192) / 4);
  k_zero<<<1024, 256, 0, stream>>>((uint32_t*)padC, (PAD31 * 2 + 8192) / 4);
  k_zero<<<1024, 256, 0, stream>>>((uint32_t*)padD, (PAD31 * 2 + 8192) / 4);
  k_zero<<<1, 256, 0, stream>>>((uint32_t*)sum, 256);  // sum+sumsq contiguous

  // weight transforms -> f16 [tap][oc][ic]
  for (int r = 0; r < 4; ++r) {
    k_wtrans<<<576, 256, 0, stream>>>(cfw[r], wgt[2 * r]);
    k_wtrans<<<576, 256, 0, stream>>>(crw[r], wgt[2 * r + 1]);
  }

  // conv1 + bn0 + relu + pool -> padA interior (63x63)
  k_conv1_stats<<<4096, 256, 0, stream>>>(X, W1, B1, sum, sumsq);
  k_finalize_stats<<<1, 128, 0, stream>>>(sum, sumsq, G0, Bt0, scale, shift,
                                          1.f / (float)(32 * 128 * 128));
  {
    size_t tot = (size_t)32 * 63 * 63 * 128;
    k_conv1_bn_pool<<<(unsigned)((tot + 255) / 256), 256, 0, stream>>>(
        X, W1, B1, scale, shift, padA);
  }

  auto conv3 = [&](const f16* inP, const f16* wg, const float* bs, const f16* rs,
                   f16* o, int H) {
    int xsegs = (H + 31) >> 5;                 // 32-pixel segments per wave
    int tiles = 32 * H * xsegs;
    dim3 grid((tiles + 3) / 4, 2, 1);
    k_conv3x3_wmma<<<grid, 128, 0, stream>>>(inP, wg, bs, rs, o, H);
  };
  auto bnblock = [&](const f16* t, const float* g_, const float* b_, f16* outP, int H) {
    size_t tot = (size_t)32 * H * H * 128;
    k_chan_stats<<<2048, 256, 0, stream>>>(t, sum, sumsq, tot);
    k_finalize_stats<<<1, 128, 0, stream>>>(sum, sumsq, g_, b_, scale, shift,
                                            1.f / (float)(32 * H * H));
    k_bn_relu_pad<<<(unsigned)((tot + 255) / 256), 256, 0, stream>>>(
        t, scale, shift, outP, H, tot);
  };
  auto rcl = [&](const f16* inP, f16* outP, f16* fb, f16* tb, int ri, int H) {
    conv3(inP, wgt[2 * ri], cfb[ri], nullptr, fb, H);       // f = conv_f(x)+b
    bnblock(fb, gg[ri], bb[ri], outP, H);                   // r = relu(bn(f))
    for (int it = 0; it < 2; ++it) {
      conv3(outP, wgt[2 * ri + 1], crb[ri], fb, tb, H);     // t = f + conv_r(r)+b
      bnblock(tb, gg[ri], bb[ri], outP, H);                 // r = relu(bn(t))
    }
  };

  rcl(padA, padB, f63, t63, 0, 63);   // rc2
  rcl(padB, padA, f63, t63, 1, 63);   // rc3
  k_maxpool<<<(3936256 + 255) / 256, 256, 0, stream>>>(padA, padC, 63, 31, 1, 1,
                                                       (size_t)3936256);
  rcl(padC, padD, f31, t31, 2, 31);   // rc4
  rcl(padD, padC, f31, t31, 3, 31);   // rc5
  k_maxpool<<<(921600 + 255) / 256, 256, 0, stream>>>(padC, p15, 31, 15, 1, 0,
                                                      (size_t)921600);
  k_gmax<<<16, 256, 0, stream>>>(p15, gm);
  k_fc<<<2, 256, 0, stream>>>(gm, FCW, FCB, OUT);
}
